// MultiHeadAttention_3676492005999
// MI455X (gfx1250) — compile-verified, matrix operand order checked by reading, not run on hardware
//
#include <hip/hip_runtime.h>
#include <hip/hip_bf16.h>
#include <math.h>

// ---------------------------------------------------------------------------
// MHA forward for MI455X (gfx1250, wave32).
// Matmuls via v_wmma_f32_16x16x32_bf16; GEMMs staged through LDS in bf16.
// ---------------------------------------------------------------------------

typedef __attribute__((ext_vector_type(16))) __bf16        v16bf;
typedef __attribute__((ext_vector_type(8)))  float         v8f;
typedef __attribute__((ext_vector_type(8)))  unsigned int  v8u;

union FragU { v8u u; v16bf bf; };

#define EMBED  1024
#define LSEQ   1024
#define NBATCH 4
#define NHEAD  16
#define HDIM   64
#define SCALE_Q 0.125f      // 64^-0.5
#define CLAMP_V 50000.0f
#define MASK_FILL -9e15f

static __device__ __forceinline__ v8f wmma_bf16(v16bf a, v16bf b, v8f c) {
    return __builtin_amdgcn_wmma_f32_16x16x32_bf16(
        false, a, false, b, (short)0, c, false, false);
}

// Pack two f32 into one dword of bf16 pair (lowers to v_cvt_pk_bf16_f32).
static __device__ __forceinline__ unsigned pack_bf16(float lo, float hi) {
    union { __bf16 b[2]; unsigned u; } r;
    r.b[0] = (__bf16)lo;
    r.b[1] = (__bf16)hi;
    return r.u;
}

// 16x32 bf16 A/B fragment from row-major f32 (K contiguous); rowptr already
// points at (row, k0 + laneHalf*8).  ISA layout: lane j<4 -> K=2j,2j+1;
// j>=4 -> K=16+2(j-4); lanes 16-31 get +8 (folded into rowptr).
static __device__ __forceinline__ v16bf load_frag_row(const float* rowptr) {
    v16bf f;
#pragma unroll
    for (int j = 0; j < 8; ++j) {
        int k = ((j >= 4) ? 16 : 0) + 2 * (j & 3);
        f[2 * j]     = (__bf16)rowptr[k];
        f[2 * j + 1] = (__bf16)rowptr[k + 1];
    }
    return f;
}

// Same fragment but K strided by `ld` floats (element k at p[k*ld]).
static __device__ __forceinline__ v16bf load_frag_strided(const float* p, int ld) {
    v16bf f;
#pragma unroll
    for (int j = 0; j < 8; ++j) {
        int k = ((j >= 4) ? 16 : 0) + 2 * (j & 3);
        f[2 * j]     = (__bf16)p[(size_t)k * ld];
        f[2 * j + 1] = (__bf16)p[(size_t)(k + 1) * ld];
    }
    return f;
}

// Fragment from LDS tile stored as packed bf16 dwords, row stride SKW dwords.
// rowp points at the lane's row; dword index laneHi*4 + {0..3} and +8.
static __device__ __forceinline__ v16bf load_frag_lds(const unsigned* rowp, int laneHi) {
    FragU f;
#pragma unroll
    for (int j = 0; j < 4; ++j) {
        f.u[j]     = rowp[laneHi * 4 + j];
        f.u[4 + j] = rowp[laneHi * 4 + 8 + j];
    }
    return f.bf;
}

// ---------------------------------------------------------------------------
// Y[M,1024] = (X[M,1024] @ W[1024,1024]^T + bias) * scale
// 128x128 tile per 256-thread workgroup; bf16 LDS staging, double buffered;
// each wave: 32x64 sub-tile = 8 accumulators, 8 WMMAs per K-step.
// ---------------------------------------------------------------------------
#define SKW 20   // LDS row stride in dwords (16 data + 4 pad -> 80B, 16B-aligned)

__global__ void __launch_bounds__(256)
gemm_bias_kernel(const float* __restrict__ X, const float* __restrict__ W,
                 const float* __restrict__ bias, float* __restrict__ Y,
                 float scale) {
    __shared__ unsigned sA[2][128][SKW];   // 20 KB : A tile 128x32 bf16
    __shared__ unsigned sB[2][128][SKW];   // 20 KB : B tile (W rows) 128x32 bf16

    const int tid  = threadIdx.x;
    const int wave = tid >> 5;
    const int lane = tid & 31;
    const int laneM  = lane & 15;
    const int laneHi = lane >> 4;

    const int bn = blockIdx.x & 7;          // N/128 = 8
    const int bm = blockIdx.x >> 3;         // M/128 = 32
    const int m0 = bm * 128, n0 = bn * 128;

    const int wm = wave >> 1;               // 0..3 : 32-row band
    const int wn = wave & 1;                // 0..1 : 64-col band

    // cooperative tile loader: thread -> (row = tid>>1, k-half = tid&1)
    const int lrow  = tid >> 1;
    const int lhalf = tid & 1;
    const float* gA = X + (size_t)(m0 + lrow) * EMBED + lhalf * 16;
    const float* gB = W + (size_t)(n0 + lrow) * EMBED + lhalf * 16;

    v8f c[2][4];
#pragma unroll
    for (int i = 0; i < 2; ++i)
#pragma unroll
        for (int j = 0; j < 4; ++j) c[i][j] = (v8f){};

    // prime buffer 0 with k0 = 0
    {
        unsigned* dA = &sA[0][lrow][lhalf * 8];
        unsigned* dB = &sB[0][lrow][lhalf * 8];
#pragma unroll
        for (int j = 0; j < 8; ++j) {
            dA[j] = pack_bf16(gA[2 * j], gA[2 * j + 1]);
            dB[j] = pack_bf16(gB[2 * j], gB[2 * j + 1]);
        }
    }

    int buf = 0;
    for (int k0 = 0; k0 < EMBED; k0 += 32, buf ^= 1) {
        __syncthreads();
        if (k0 + 32 < EMBED) {                      // prefetch next K tile
            const float* pA = gA + k0 + 32;
            const float* pB = gB + k0 + 32;
            unsigned* dA = &sA[buf ^ 1][lrow][lhalf * 8];
            unsigned* dB = &sB[buf ^ 1][lrow][lhalf * 8];
#pragma unroll
            for (int j = 0; j < 8; ++j) {
                dA[j] = pack_bf16(pA[2 * j], pA[2 * j + 1]);
                dB[j] = pack_bf16(pB[2 * j], pB[2 * j + 1]);
            }
        }
        // fragments from LDS
        v16bf af[2], bfg[4];
#pragma unroll
        for (int i = 0; i < 2; ++i)
            af[i] = load_frag_lds(&sA[buf][wm * 32 + i * 16 + laneM][0], laneHi);
#pragma unroll
        for (int j = 0; j < 4; ++j)
            bfg[j] = load_frag_lds(&sB[buf][wn * 64 + j * 16 + laneM][0], laneHi);
#pragma unroll
        for (int i = 0; i < 2; ++i)
#pragma unroll
            for (int j = 0; j < 4; ++j)
                c[i][j] = wmma_bf16(af[i], bfg[j], c[i][j]);
    }

    // epilogue: (acc + bias) * scale
#pragma unroll
    for (int j = 0; j < 4; ++j) {
        const int n = n0 + wn * 64 + j * 16 + laneM;
        const float bv = bias[n];
#pragma unroll
        for (int i = 0; i < 2; ++i) {
#pragma unroll
            for (int r = 0; r < 8; ++r) {
                const int m = m0 + wm * 32 + i * 16 + r + laneHi * 8;
                Y[(size_t)m * EMBED + n] = (c[i][j][r] + bv) * scale;
            }
        }
    }
}

// ---------------------------------------------------------------------------
// Fused attention: one block = one (b, h, 16-query tile), 8 waves.
// Phase 1: logits = Q K^T (WMMA, Q frags hoisted), clamp+mask -> 64KB LDS.
// Phase 2: softmax (wave per 2 rows, shfl_xor), write attn probs to global.
// Phase 3: ctx = attn @ V, split-K across wave pairs + LDS reduction.
// ---------------------------------------------------------------------------
__global__ void __launch_bounds__(256)
attn_kernel(const float* __restrict__ Qp, const float* __restrict__ Kp,
            const float* __restrict__ Vp, const int* __restrict__ mask,
            float* __restrict__ attn_out, float* __restrict__ ctx) {
    __shared__ float slog[16][LSEQ];                   // 64 KB logits tile

    const int blk = blockIdx.x;                        // b*H*64 + h*64 + qt
    const int qt = blk & 63;
    const int h  = (blk >> 6) & 15;
    const int b  = blk >> 10;
    const int tid  = threadIdx.x;
    const int wave = tid >> 5;
    const int lane = tid & 31;
    const int laneM  = lane & 15;
    const int laneHi = lane >> 4;
    const int q0 = qt * 16;

    const float* Qb = Qp + (size_t)b * LSEQ * EMBED + (size_t)h * HDIM;
    const float* Kb = Kp + (size_t)b * LSEQ * EMBED + (size_t)h * HDIM;
    const float* Vb = Vp + (size_t)b * LSEQ * EMBED + (size_t)h * HDIM;

    // ---- Phase 1: logits; Q fragments loaded once, reused for 8 key tiles
    const float* arow = Qb + (size_t)(q0 + laneM) * EMBED + laneHi * 8;
    v16bf aq0 = load_frag_row(arow);
    v16bf aq1 = load_frag_row(arow + 32);
#pragma unroll
    for (int t = 0; t < 8; ++t) {
        const int n0 = wave * 128 + t * 16;            // key tile base
        const float* brow = Kb + (size_t)(n0 + laneM) * EMBED + laneHi * 8;
        v8f c = {};
        c = wmma_bf16(aq0, load_frag_row(brow), c);
        c = wmma_bf16(aq1, load_frag_row(brow + 32), c);
        const int col = n0 + laneM;
        const int mv  = mask[b * LSEQ + col];
        const float madd = (mv == 0) ? MASK_FILL : (float)mv;
#pragma unroll
        for (int r = 0; r < 8; ++r) {
            float x = fminf(fmaxf(c[r], -CLAMP_V), CLAMP_V);
            slog[r + laneHi * 8][col] = x + madd;
        }
    }
    __syncthreads();

    // ---- Phase 2: softmax, wave w owns rows 2w,2w+1; lane owns 32 cols
#pragma unroll
    for (int rr = 0; rr < 2; ++rr) {
        const int row = wave * 2 + rr;
        float* rowp = slog[row];
        const int c0 = lane * 32;
        float mx = -3.0e38f;
#pragma unroll 8
        for (int j = 0; j < 32; ++j) mx = fmaxf(mx, rowp[c0 + j]);
#pragma unroll
        for (int o = 16; o > 0; o >>= 1) mx = fmaxf(mx, __shfl_xor(mx, o, 32));
        float s = 0.0f;
#pragma unroll 8
        for (int j = 0; j < 32; ++j) {
            float e = __expf(rowp[c0 + j] - mx);
            rowp[c0 + j] = e;
            s += e;
        }
#pragma unroll
        for (int o = 16; o > 0; o >>= 1) s += __shfl_xor(s, o, 32);
        const float inv = 1.0f / s;
        float* ao = attn_out +
            (((size_t)(b * NHEAD + h) * LSEQ) + q0 + row) * LSEQ + c0;
#pragma unroll 8
        for (int j = 0; j < 32; ++j) {
            float p = rowp[c0 + j] * inv;
            rowp[c0 + j] = p;
            ao[j] = p;
        }
    }
    __syncthreads();

    // ---- Phase 3: ctx = attn @ V; all 8 waves: dtile = wave&3, K-half = wave>>2
    {
        const int n0    = (wave & 3) * 16;             // d-column tile
        const int kbase = (wave >> 2) * 512;           // key range half
        v8f c = {};
        for (int k0 = kbase; k0 < kbase + 512; k0 += 32) {
            v16bf a = load_frag_row(&slog[laneM][k0 + laneHi * 8]);
            const float* vp = Vb + (size_t)(k0 + laneHi * 8) * EMBED + n0 + laneM;
            c = wmma_bf16(a, load_frag_strided(vp, EMBED), c);
        }
        __syncthreads();                               // slog reads done
        float* scratch = &slog[0][0];                  // reuse as 4x16x16 partials
        const int idx = (wave & 3) * 256 + laneHi * 128; // + r*16 + laneM below
        if (wave >= 4) {
#pragma unroll
            for (int r = 0; r < 8; ++r)
                scratch[idx + r * 16 + laneM] = c[r];
        }
        __syncthreads();
        if (wave < 4) {
#pragma unroll
            for (int r = 0; r < 8; ++r) {
                const float sum = c[r] + scratch[idx + r * 16 + laneM];
                const int q = q0 + r + laneHi * 8;
                ctx[((size_t)b * LSEQ + q) * EMBED + h * HDIM + n0 + laneM] = sum;
            }
        }
    }
}

// ---------------------------------------------------------------------------
extern "C" void kernel_launch(void* const* d_in, const int* in_sizes, int n_in,
                              void* d_out, int out_size, void* d_ws, size_t ws_size,
                              hipStream_t stream) {
    const float* q    = (const float*)d_in[0];
    const float* k    = (const float*)d_in[1];
    const float* v    = (const float*)d_in[2];
    const int*   mask = (const int*)d_in[3];
    const float* Wq = (const float*)d_in[4];
    const float* bq = (const float*)d_in[5];
    const float* Wk = (const float*)d_in[6];
    const float* bk = (const float*)d_in[7];
    const float* Wv = (const float*)d_in[8];
    const float* bv = (const float*)d_in[9];
    const float* Wo = (const float*)d_in[10];
    const float* bo = (const float*)d_in[11];

    float* out  = (float*)d_out;                                   // [4,1024,1024]
    float* attn = out + (size_t)NBATCH * LSEQ * EMBED;             // [4,16,1024,1024]

    const size_t mat = (size_t)NBATCH * LSEQ * EMBED;              // 4M floats
    float* Qp  = (float*)d_ws;
    float* Kp  = Qp + mat;
    float* Vp  = Kp + mat;
    float* ctx = Vp + mat;                                         // 64 MB total

    const int gemmBlocks = (NBATCH * LSEQ / 128) * (EMBED / 128);  // 32*8 = 256

    gemm_bias_kernel<<<gemmBlocks, 256, 0, stream>>>(q, Wq, bq, Qp, SCALE_Q);
    gemm_bias_kernel<<<gemmBlocks, 256, 0, stream>>>(k, Wk, bk, Kp, 1.0f);
    gemm_bias_kernel<<<gemmBlocks, 256, 0, stream>>>(v, Wv, bv, Vp, 1.0f);

    const int attnBlocks = NBATCH * NHEAD * (LSEQ / 16);           // 4096
    attn_kernel<<<attnBlocks, 256, 0, stream>>>(Qp, Kp, Vp, mask, attn, ctx);

    gemm_bias_kernel<<<gemmBlocks, 256, 0, stream>>>(ctx, Wo, bo, out, 1.0f);
}
